// NeuralODE_1176821039669
// MI455X (gfx1250) — matrix operand order repaired and checked by reading, not compiled, not run on hardware
//
#include <hip/hip_runtime.h>
#include <hip/hip_bf16.h>
#include <stdint.h>

// ---------------------------------------------------------------------------
// Types
// ---------------------------------------------------------------------------
typedef _Float16 half8  __attribute__((ext_vector_type(8)));
typedef _Float16 half16 __attribute__((ext_vector_type(16)));
typedef float    float8 __attribute__((ext_vector_type(8)));
typedef unsigned int u32x4 __attribute__((ext_vector_type(4)));
typedef int          i32x8 __attribute__((ext_vector_type(8)));
typedef int          i32x4 __attribute__((ext_vector_type(4)));

// ---------------------------------------------------------------------------
// Tsit5 tableau / controller constants (match reference)
// ---------------------------------------------------------------------------
#define C2v 0.161f
#define C3v 0.327f
#define C4v 0.9f
#define C5v 0.9800255409045097f
#define A21v 0.161f
#define A31v (-0.008480655492356989f)
#define A32v 0.335480655492357f
#define A41v 2.8971530571054935f
#define A42v (-6.359448489975075f)
#define A43v 4.3622954328695815f
#define A51v 5.325864828439257f
#define A52v (-11.748883564062828f)
#define A53v 7.4955393428898365f
#define A54v (-0.09249506636175525f)
#define A61v 5.86145544294642f
#define A62v (-12.92096931784711f)
#define A63v 8.159367898576159f
#define A64v (-0.071584973281401f)
#define A65v (-0.028269050394068383f)
#define B1v 0.09646076681806523f
#define B2v 0.01f
#define B3v 0.4798896504144996f
#define B4v 1.379008574103742f
#define B5v (-3.290069515436081f)
#define B6v 2.324710524099774f
#define E1v (-0.001780011052226f)
#define E2v (-0.000816434459657f)
#define E3v 0.007880878010262f
#define E4v (-0.144711007173263f)
#define E5v 0.582357165452555f
#define E6v (-0.458082105929187f)
#define E7v 0.015151515151515152f

// ---------------------------------------------------------------------------
// LDS layout (bytes). Packed f16 weights first (so TDM lds_addr == 0).
//   W1 frags: 4kt x 16nt  -> 32768 halfs
//   W2 frags: 8kt x 16nt  -> 65536 halfs
//   W3 frags: 8kt x  8nt  -> 32768 halfs
// ---------------------------------------------------------------------------
enum : unsigned {
  W1H = 0, W2H = 32768, W3H = 98304,           // half offsets inside packed blob
  WPACK_BYTES = 262144,
  S0 = 136,                                     // act0 row stride (halfs)
  S1 = 264,                                     // act1/act2 row stride (halfs)
  ACT0_OFF = WPACK_BYTES,                       // 16 x 136 halfs
  ACT1_OFF = ACT0_OFF + 16 * S0 * 2,
  ACT2_OFF = ACT1_OFF + 16 * S1 * 2,
  B1_OFF   = ACT2_OFF + 16 * S1 * 2,            // 256 f32
  W1T_OFF  = B1_OFF + 1024,                     // 256 f32 (t-row of W1)
  B2_OFF   = W1T_OFF + 1024,                    // 256 f32
  B3_OFF   = B2_OFF + 1024,                     // 128 f32
  TS_OFF   = B3_OFF + 512,                      // 16 f32
  TCUR_OFF = TS_OFF + 64,
  DT_OFF   = TCUR_OFF + 64,
  DTC_OFF  = DT_OFF + 64,
  TSTG_OFF = DTC_OFF + 64,
  ERR_OFF  = TSTG_OFF + 64,
  MASK_OFF = ERR_OFF + 64,
  SMEM_BYTES = MASK_OFF + 16                    // ~287.4 KB < 320 KB WGP LDS
};

// ---------------------------------------------------------------------------
// Kernel 0: re-pack fp32 weights into f16 WMMA B-fragment byte order.
// For weight W[K][N]: fragment id (kt,ntl); lane L holds 16 contiguous halfs
//   n = ntl*16 + (L&15),  K = kt*32 + (L>>4)*16 + j   (j = 0..15)
// so the main kernel fetches each B operand as two ds_load_b128.
// ---------------------------------------------------------------------------
__device__ inline void pack_one(const float* W, int N, int NT, int e, _Float16* out) {
  const int j = e & 15;
  const int c = e >> 4;
  const int lane = c & 31;
  const int fid = c >> 5;
  const int ntl = fid % NT;
  const int kt = fid / NT;
  const int n = ntl * 16 + (lane & 15);
  const int k = kt * 32 + (lane >> 4) * 16 + j;
  out[e] = (_Float16)W[k * N + n];
}

__global__ __launch_bounds__(256) void pack_weights(const float* __restrict__ W1,
                                                    const float* __restrict__ W2,
                                                    const float* __restrict__ W3,
                                                    _Float16* __restrict__ wp) {
  const int e = blockIdx.x * 256 + threadIdx.x;
  if (e < 32768) {                         // W1: K=128 (t-row handled separately), N=256
    pack_one(W1, 256, 16, e, wp + W1H);
  } else if (e < 98304) {                  // W2: 256x256
    pack_one(W2, 256, 16, e - 32768, wp + W2H);
  } else if (e < 131072) {                 // W3: 256x128
    pack_one(W3, 128, 8, e - 98304, wp + W3H);
  }
}

// ---------------------------------------------------------------------------
// One 16x16 output tile of A[16,K] @ B[K,16], K = KT*32, f16 in / f32 acc.
// A frag: row m=ln, K-chunk (hi*16 .. +15) per k-tile.
// B frag: pre-packed, lane-contiguous 32 B.
// ---------------------------------------------------------------------------
template <int KT, int NT>
__device__ inline void gemm_tile(const _Float16* act, int strideH,
                                 const _Float16* wp, int ntl, int lane, float8& acc) {
  const int ln = lane & 15, hi = lane >> 4;
#pragma unroll
  for (int kt = 0; kt < KT; ++kt) {
    union { half16 v; half8 h[2]; } A, B;
    const _Float16* pa = act + ln * strideH + kt * 32 + hi * 16;
    A.h[0] = *(const half8*)pa;
    A.h[1] = *(const half8*)(pa + 8);
    const _Float16* pb = wp + ((kt * NT + ntl) * 32 + lane) * 16;
    B.h[0] = *(const half8*)pb;
    B.h[1] = *(const half8*)(pb + 8);
    acc = __builtin_amdgcn_wmma_f32_16x16x32_f16(false, A.v, false, B.v,
                                                 (short)0, acc, false, false);
  }
}

// ---------------------------------------------------------------------------
// f(t, y) for the 16-sample tile: act0 (f16) and TSTG already populated.
// GEMM3's C-fragment layout == per-thread state ownership, so kout lands
// directly in the stage registers.
// ---------------------------------------------------------------------------
__device__ inline void mlp_eval(char* smem, int wave, int lane, float kout[8]) {
  const int ln = lane & 15, hi = lane >> 4;
  _Float16* wp = (_Float16*)smem;
  _Float16* a0 = (_Float16*)(smem + ACT0_OFF);
  _Float16* a1 = (_Float16*)(smem + ACT1_OFF);
  _Float16* a2 = (_Float16*)(smem + ACT2_OFF);
  const float* b1f = (const float*)(smem + B1_OFF);
  const float* w1t = (const float*)(smem + W1T_OFF);
  const float* b2f = (const float*)(smem + B2_OFF);
  const float* b3f = (const float*)(smem + B3_OFF);
  const float* tst = (const float*)(smem + TSTG_OFF);

  // layer 1: [16,128] @ [128,256] + t * W1[128,:] + b1, swish
#pragma unroll
  for (int q = 0; q < 2; ++q) {
    const int ntl = wave * 2 + q;
    float8 acc = {};
    gemm_tile<4, 16>(a0, S0, wp + W1H, ntl, lane, acc);
#pragma unroll
    for (int r = 0; r < 8; ++r) {
      const int m = r + hi * 8, n = ntl * 16 + ln;
      float v = acc[r] + b1f[n] + tst[m] * w1t[n];
      a1[m * S1 + n] = (_Float16)(v / (1.0f + __expf(-v)));
    }
  }
  __syncthreads();
  // layer 2: [16,256] @ [256,256] + b2, swish
#pragma unroll
  for (int q = 0; q < 2; ++q) {
    const int ntl = wave * 2 + q;
    float8 acc = {};
    gemm_tile<8, 16>(a1, S1, wp + W2H, ntl, lane, acc);
#pragma unroll
    for (int r = 0; r < 8; ++r) {
      const int m = r + hi * 8, n = ntl * 16 + ln;
      float v = acc[r] + b2f[n];
      a2[m * S1 + n] = (_Float16)(v / (1.0f + __expf(-v)));
    }
  }
  __syncthreads();
  // layer 3: [16,256] @ [256,128] + b3 -> stage derivative, in registers
  {
    float8 acc = {};
    gemm_tile<8, 8>(a2, S1, wp + W3H, wave, lane, acc);
#pragma unroll
    for (int r = 0; r < 8; ++r) kout[r] = acc[r] + b3f[wave * 16 + ln];
  }
  __syncthreads();
}

// Write stage input x (per-thread f32) into act0 as f16; lanes<16 of wave 0
// publish per-sample stage time t_m + c*dtc_m.
__device__ inline void store_stage(char* smem, int tid, int lane, int wave,
                                   const float x[8], float cval) {
  const int ln = lane & 15, hi = lane >> 4;
  _Float16* a0 = (_Float16*)(smem + ACT0_OFF);
  float* tst = (float*)(smem + TSTG_OFF);
  const float* tcur = (const float*)(smem + TCUR_OFF);
  const float* dtc = (const float*)(smem + DTC_OFF);
  const int n = wave * 16 + ln;
#pragma unroll
  for (int r = 0; r < 8; ++r) a0[(r + hi * 8) * S0 + n] = (_Float16)x[r];
  if (tid < 16) tst[tid] = tcur[tid] + cval * dtc[tid];
  __syncthreads();
}

// ---------------------------------------------------------------------------
// Main kernel: one block integrates 16 samples through all 15 intervals.
// ---------------------------------------------------------------------------
__global__ __launch_bounds__(256, 1)
void node_tsit5(const float* __restrict__ y0g, const float* __restrict__ tsg,
                const float* __restrict__ W1g, const float* __restrict__ b1g,
                const float* __restrict__ b2g, const float* __restrict__ b3g,
                const _Float16* __restrict__ wpg, float* __restrict__ outg) {
  extern __shared__ char smem[];
  const int tid = threadIdx.x;
  const int lane = tid & 31, wave = tid >> 5;
  const int ln = lane & 15, hi = lane >> 4;
  const int n_own = wave * 16 + ln;
  const int mbase = blockIdx.x * 16;

  // ---- weights -> LDS -------------------------------------------------
#ifdef __has_builtin
#if __has_builtin(__builtin_amdgcn_tensor_load_to_lds)
  if (wave == 0) {
    // D# per CDNA5 ISA ch.8: 2-D tensor of 8192x8 dwords (= 256 KB),
    // data_size=4B, tile == tensor, lds_addr = 0 (packed weights region).
    const uint64_t ga = (uint64_t)(uintptr_t)wpg;
    u32x4 g0;
    g0[0] = 1u;                                   // count = 1 (valid)
    g0[1] = 0u;                                   // lds_addr = 0
    g0[2] = (unsigned)(ga & 0xFFFFFFFFu);         // global_addr[31:0]
    g0[3] = (unsigned)((ga >> 32) & 0x1FFFFFFu) | (2u << 30);  // addr hi | type=2
    i32x8 g1;
    g1[0] = (int)(2u << 16);                      // data_size = 4B
    g1[1] = (int)(8192u << 16);                   // tensor_dim0[15:0] @bit48
    g1[2] = (int)(8u << 16);                      // tensor_dim1 = 8 @bit80
    g1[3] = (int)(8192u << 16);                   // tile_dim0 @bit112
    g1[4] = 8;                                    // tile_dim1 = 8
    g1[5] = 8192;                                 // tensor_dim0_stride
    g1[6] = 0;
    g1[7] = 0;
    i32x4 z4 = {0, 0, 0, 0};
    i32x8 z8 = {0, 0, 0, 0, 0, 0, 0, 0};
    __builtin_amdgcn_tensor_load_to_lds(g0, g1, z4, z4, z8, 0);
    __builtin_amdgcn_s_wait_tensorcnt(0);
  }
#else
  {
    const uint4* src = (const uint4*)wpg;
    uint4* dst = (uint4*)smem;
    for (int i = tid; i < (int)(WPACK_BYTES / 16); i += 256) dst[i] = src[i];
  }
#endif
#else
  {
    const uint4* src = (const uint4*)wpg;
    uint4* dst = (uint4*)smem;
    for (int i = tid; i < (int)(WPACK_BYTES / 16); i += 256) dst[i] = src[i];
  }
#endif

  float* b1f = (float*)(smem + B1_OFF);
  float* w1t = (float*)(smem + W1T_OFF);
  float* b2f = (float*)(smem + B2_OFF);
  float* b3f = (float*)(smem + B3_OFF);
  float* tsl = (float*)(smem + TS_OFF);
  float* tcur = (float*)(smem + TCUR_OFF);
  float* dtl = (float*)(smem + DT_OFF);
  float* dtcf = (float*)(smem + DTC_OFF);
  float* errb = (float*)(smem + ERR_OFF);
  unsigned* maska = (unsigned*)(smem + MASK_OFF);

  b1f[tid] = b1g[tid];
  w1t[tid] = W1g[128 * 256 + tid];   // t-row of W1
  b2f[tid] = b2g[tid];
  if (tid < 128) b3f[tid] = b3g[tid];
  if (tid < 16) tsl[tid] = tsg[tid];
  __syncthreads();

  // ---- per-thread state (WMMA C-fragment ownership) -------------------
  float y[8];
#pragma unroll
  for (int r = 0; r < 8; ++r) {
    const int m = r + hi * 8;
    y[r] = y0g[(size_t)(mbase + m) * 128 + n_own];
    outg[((size_t)(mbase + m) * 16 + 0) * 128 + n_own] = y[r];  // save t0 row
  }
  if (tid < 16) {
    tcur[tid] = tsl[0];
    dtl[tid] = tsl[1] - tsl[0];
    dtcf[tid] = 0.0f;
  }
  __syncthreads();

  float kk[7][8];
  // initial k1 = f(t0, y0)
  {
    float x[8];
#pragma unroll
    for (int r = 0; r < 8; ++r) x[r] = y[r];
    store_stage(smem, tid, lane, wave, x, 0.0f);
    mlp_eval(smem, wave, lane, kk[0]);
  }

  for (int it = 1; it < 16; ++it) {
    const float t1 = tsl[it];
    for (int st = 0; st < 4; ++st) {
      if (tid < 16) {
        dtcf[tid] = fminf(dtl[tid], t1 - tcur[tid]);
        errb[tid] = 0.0f;
      }
      if (tid == 0) *maska = 0u;
      __syncthreads();

      float dr[8];
#pragma unroll
      for (int r = 0; r < 8; ++r) dr[r] = dtcf[r + hi * 8];

      constexpr float Cc[5] = {C2v, C3v, C4v, C5v, 1.0f};
      constexpr float Ac[5][5] = {
          {A21v, 0, 0, 0, 0},
          {A31v, A32v, 0, 0, 0},
          {A41v, A42v, A43v, 0, 0},
          {A51v, A52v, A53v, A54v, 0},
          {A61v, A62v, A63v, A64v, A65v}};
#pragma unroll
      for (int s = 0; s < 5; ++s) {     // stages k2..k6
        float x[8];
#pragma unroll
        for (int r = 0; r < 8; ++r) {
          float a = 0.0f;
#pragma unroll
          for (int j = 0; j <= s; ++j) a += Ac[s][j] * kk[j][r];
          x[r] = y[r] + dr[r] * a;
        }
        store_stage(smem, tid, lane, wave, x, Cc[s]);
        mlp_eval(smem, wave, lane, kk[s + 1]);
      }

      float y1[8];
#pragma unroll
      for (int r = 0; r < 8; ++r)
        y1[r] = y[r] + dr[r] * (B1v * kk[0][r] + B2v * kk[1][r] + B3v * kk[2][r] +
                                B4v * kk[3][r] + B5v * kk[4][r] + B6v * kk[5][r]);
      store_stage(smem, tid, lane, wave, y1, 1.0f);
      mlp_eval(smem, wave, lane, kk[6]);  // FSAL k7

      // error norm: sum (err/scale)^2 over the 128 dims of each sample
#pragma unroll
      for (int r = 0; r < 8; ++r) {
        const float e = dr[r] * (E1v * kk[0][r] + E2v * kk[1][r] + E3v * kk[2][r] +
                                 E4v * kk[3][r] + E5v * kk[4][r] + E6v * kk[5][r] +
                                 E7v * kk[6][r]);
        const float sc = 1e-6f + 1e-3f * fmaxf(fabsf(y[r]), fabsf(y1[r]));
        const float q = e / sc;
        atomicAdd(&errb[r + hi * 8], q * q);   // ds_add_f32
      }
      __syncthreads();

      // per-sample I-controller (lanes 0..15 of wave 0)
      if (tid < 16) {
        const float t = tcur[tid], d = dtl[tid], dc = dtcf[tid];
        const bool done = t >= t1 - 1e-8f;
        const float en = sqrtf(errb[tid] * (1.0f / 128.0f));
        const bool acc = en <= 1.0f;
        float f = 0.9f * __powf(en > 0.0f ? en : 1.0f, -0.2f);
        f = fminf(fmaxf(f, 0.2f), 10.0f);
        tcur[tid] = done ? t : (acc ? t + dc : t);
        dtl[tid] = done ? d : dc * f;
        if (acc && !done) atomicOr(maska, 1u << tid);
      }
      __syncthreads();
      const unsigned mk = *maska;
#pragma unroll
      for (int r = 0; r < 8; ++r) {
        const int m = r + hi * 8;
        if ((mk >> m) & 1u) {
          y[r] = y1[r];
          kk[0][r] = kk[6][r];  // FSAL
        }
      }
      __syncthreads();
    }
    // save interval endpoint
#pragma unroll
    for (int r = 0; r < 8; ++r) {
      const int m = r + hi * 8;
      outg[((size_t)(mbase + m) * 16 + it) * 128 + n_own] = y[r];
    }
  }
}

// ---------------------------------------------------------------------------
// Host entry
// ---------------------------------------------------------------------------
extern "C" void kernel_launch(void* const* d_in, const int* in_sizes, int n_in,
                              void* d_out, int out_size, void* d_ws, size_t ws_size,
                              hipStream_t stream) {
  (void)in_sizes; (void)n_in; (void)out_size; (void)ws_size;
  const float* y0 = (const float*)d_in[0];
  const float* ts = (const float*)d_in[1];
  const float* W1 = (const float*)d_in[2];
  const float* b1 = (const float*)d_in[3];
  const float* W2 = (const float*)d_in[4];
  const float* b2 = (const float*)d_in[5];
  const float* W3 = (const float*)d_in[6];
  const float* b3 = (const float*)d_in[7];
  _Float16* wp = (_Float16*)d_ws;   // 256 KB packed f16 weight image

  pack_weights<<<512, 256, 0, stream>>>(W1, W2, W3, wp);

  (void)hipFuncSetAttribute((const void*)node_tsit5,
                            hipFuncAttributeMaxDynamicSharedMemorySize,
                            (int)SMEM_BYTES);
  node_tsit5<<<32, 256, SMEM_BYTES, stream>>>(y0, ts, W1, b1, b2, b3, wp,
                                              (float*)d_out);
}
